// FeatureAggregator_31310311588319
// MI455X (gfx1250) — matrix operand order repaired and checked by reading, not compiled, not run on hardware
//
#include <hip/hip_runtime.h>
#include <hip/hip_bf16.h>
#include <stdint.h>

// ---------------------------------------------------------------------------
// Problem constants (match the reference)
// ---------------------------------------------------------------------------
constexpr int B_   = 2;
constexpr int NQ   = 4096;
constexpr int NA   = 2048;
constexpr int DIM  = 512;   // == DIM_INP == HID
constexpr int HID  = 512;
constexpr int NBLK = 5;
constexpr int KNN  = 8;

typedef __attribute__((ext_vector_type(16))) __bf16 bf16x16;
typedef __attribute__((ext_vector_type(8)))  float  f32x8;
typedef unsigned int v4u __attribute__((ext_vector_type(4)));
typedef int          v8i __attribute__((ext_vector_type(8)));
typedef int          v4i __attribute__((ext_vector_type(4)));

// ---------------------------------------------------------------------------
// Generic WMMA GEMM:  C[M,N] = op(A[M,K]) * Bt[N,K]^T (+ bias) (+ Cin) (ReLU?)
//   Bt: pre-converted bf16 weights, TRANSPOSED to [N,K] row-major.
//   AMODE: 0 = fp32 A, 1 = fp32 A with ReLU applied on load, 2 = bf16 A
//   ACC:   add Cin[M,N] (residual accumulate)
// Block tile 128x64, 8 waves (wave32), each wave 32x32 via 2x2 wmma 16x16x32.
// B panel (64 x K bf16 = 64KB) loaded ONCE per block by the Tensor Data Mover
// with 16B/row LDS padding (bank-conflict-free fragment reads); only the A
// tile is re-staged per K-step.
// ---------------------------------------------------------------------------
constexpr int BM = 128, BN = 64, BK = 32, BKP = 48;
constexpr int BROW = 1040;                 // 512*2 + 16B TDM pad per row
constexpr int SA_BYTES = BM * BKP * 2;     // 12288

template <int AMODE, bool ACC, bool RELU_OUT, bool OUT_BF16>
__global__ __launch_bounds__(256) void gemm_bf16_wmma(
    const void* __restrict__ Aall, const __bf16* __restrict__ Bt,
    const float* __restrict__ bias, const float* __restrict__ Cin,
    float* __restrict__ CoutF, __bf16* __restrict__ CoutB,
    int M, int N, int K)
{
  __shared__ __align__(16) char smem[SA_BYTES + BN * BROW];
  __bf16 (*sA)[BKP] = reinterpret_cast<__bf16 (*)[BKP]>(smem);
  char* sB = smem + SA_BYTES;

  const int tid  = threadIdx.x;
  const int m0   = blockIdx.y * BM;
  const int n0   = blockIdx.x * BN;
  const int wid  = tid >> 5;
  const int lane = tid & 31;
  const int wm   = wid & 3;       // 4 waves along M
  const int wn   = wid >> 2;      // 2 waves along N
  const int half = lane >> 4;
  const int l16  = lane & 15;

  // ---- one-shot B panel load: Bt[n0 .. n0+63][0 .. K-1] -> LDS (padded) ----
#if defined(__HIP_DEVICE_COMPILE__) && __has_builtin(__builtin_amdgcn_tensor_load_to_lds)
  if (wid == 0) {
    const unsigned long long ga =
        (unsigned long long)(uintptr_t)(Bt + (size_t)n0 * K);
    v4u g0;
    g0[0] = 1u;                                   // count=1, user descriptor
    g0[1] = (unsigned)(uintptr_t)sB;              // LDS byte address
    g0[2] = (unsigned)ga;                         // global_addr[31:0]
    g0[3] = (unsigned)((ga >> 32) & 0x1FFFFFFull) // global_addr[56:32]
          | (2u << 30);                           // type = 2 ("image")
    v8i g1;
    g1[0] = (int)((1u << 16)        // data_size = 2 bytes
                | (1u << 20)        // pad_enable
                | (7u << 22)        // pad_interval: 256 DWORDs (one 1024B row)
                | (3u << 25));      // pad_amount: 4 DWORDs (16B)
    g1[1] = (int)((unsigned)K << 16);          // tensor_dim0[15:0] = 512
    g1[2] = (int)(((unsigned)BN) << 16);       // dim0 hi=0 | tensor_dim1 lo = 64
    g1[3] = (int)((unsigned)K << 16);          // dim1 hi=0 | tile_dim0 = 512
    g1[4] = (int)BN;                           // tile_dim1 = 64, tile_dim2 = 0
    g1[5] = (int)K;                            // tensor_dim0_stride lo32 = 512
    g1[6] = 0;                                 // stride hi | dim1_stride lo
    g1[7] = 0;
    v4i z4 = {0, 0, 0, 0};
    v8i z8 = {0, 0, 0, 0, 0, 0, 0, 0};
    __builtin_amdgcn_tensor_load_to_lds(g0, g1, z4, z4, z8, 0);
    __builtin_amdgcn_s_wait_tensorcnt(0);
  }
#else
  {  // cooperative fallback: 4 threads per row, 128 bf16 each
    const int r   = tid >> 2;
    const int seg = (tid & 3) * 128;
    const __bf16* s = Bt + (size_t)(n0 + r) * K + seg;
    __bf16*       d = (__bf16*)(sB + (size_t)r * BROW) + seg;
    #pragma unroll
    for (int i = 0; i < 128; ++i) d[i] = s[i];
  }
#endif
  __syncthreads();

  f32x8 acc[2][2] = {};

  const int ar = tid >> 1;        // 0..127 : A row in tile
  const int ac = (tid & 1) * 16;  // 0/16   : A col base

  for (int kb = 0; kb < K; kb += BK) {
    // ---- stage A tile (convert fp32 -> bf16 / fused ReLU if needed) ----
    if (AMODE == 2) {
      const __bf16* src = (const __bf16*)Aall + (size_t)(m0 + ar) * K + kb + ac;
      #pragma unroll
      for (int i = 0; i < 16; ++i) sA[ar][ac + i] = src[i];
    } else {
      const float* src = (const float*)Aall + (size_t)(m0 + ar) * K + kb + ac;
      #pragma unroll
      for (int i = 0; i < 16; ++i) {
        float v = src[i];
        if (AMODE == 1) v = fmaxf(v, 0.0f);
        sA[ar][ac + i] = (__bf16)v;
      }
    }
    // software prefetch of the next K-slab of A (global_prefetch_b8)
    if (kb + BK < K) {
      if (AMODE == 2)
        __builtin_prefetch((const __bf16*)Aall + (size_t)(m0 + ar) * K + kb + BK + ac, 0, 0);
      else
        __builtin_prefetch((const float*)Aall + (size_t)(m0 + ar) * K + kb + BK + ac, 0, 0);
    }
    __syncthreads();

    // ---- fragments: contiguous k = half*16 + 0..15 for A and B ----
    bf16x16 afrag[2], bfrag[2];
    #pragma unroll
    for (int mf = 0; mf < 2; ++mf) {
      const __bf16* ap = &sA[wm * 32 + mf * 16 + l16][0];
      #pragma unroll
      for (int i = 0; i < 8; ++i) {
        afrag[mf][i]     = ap[half * 16 + i];      // K = half*16 + 0..7
        afrag[mf][8 + i] = ap[half * 16 + 8 + i];  // K = half*16 + 8..15
      }
    }
    #pragma unroll
    for (int nf = 0; nf < 2; ++nf) {
      const __bf16* bp =
          (const __bf16*)(sB + (size_t)(wn * 32 + nf * 16 + l16) * BROW)
          + kb + half * 16;                        // K = half*16 + 0..15
      #pragma unroll
      for (int i = 0; i < 16; ++i) bfrag[nf][i] = bp[i];
    }
    #pragma unroll
    for (int mf = 0; mf < 2; ++mf)
      #pragma unroll
      for (int nf = 0; nf < 2; ++nf)
        acc[mf][nf] = __builtin_amdgcn_wmma_f32_16x16x32_bf16(
            false, afrag[mf], false, bfrag[nf], (short)0, acc[mf][nf], false, false);
    __syncthreads();
  }

  // ---- fused epilogue ----
  #pragma unroll
  for (int mf = 0; mf < 2; ++mf) {
    #pragma unroll
    for (int nf = 0; nf < 2; ++nf) {
      const int gn = n0 + wn * 32 + nf * 16 + l16;
      #pragma unroll
      for (int i = 0; i < 8; ++i) {
        const int gm = m0 + wm * 32 + mf * 16 + half * 8 + i;
        float v = acc[mf][nf][i];
        if (bias) v += bias[gn];
        if (ACC)  v += Cin[(size_t)gm * N + gn];
        if (RELU_OUT) v = fmaxf(v, 0.0f);
        if (OUT_BF16) CoutB[(size_t)gm * N + gn] = (__bf16)v;
        else          CoutF[(size_t)gm * N + gn] = v;
      }
    }
  }
}

// ---------------------------------------------------------------------------
// One-time weight prep: fp32 [K,N] -> bf16 [N,K] (transpose via LDS tile)
// ---------------------------------------------------------------------------
__global__ __launch_bounds__(256) void wtrans_kernel(
    const float* __restrict__ src, __bf16* __restrict__ dst)
{
  __shared__ float t[32][33];
  const int mz = blockIdx.z;
  const float* S = src + (size_t)mz * DIM * DIM;
  __bf16*      D = dst + (size_t)mz * DIM * DIM;
  const int x0 = blockIdx.x * 32;   // n
  const int y0 = blockIdx.y * 32;   // k
  const int tx = threadIdx.x & 31, ty = threadIdx.x >> 5;
  #pragma unroll
  for (int i = 0; i < 32; i += 8)
    t[ty + i][tx] = S[(size_t)(y0 + ty + i) * DIM + (x0 + tx)];
  __syncthreads();
  #pragma unroll
  for (int i = 0; i < 32; i += 8)
    D[(size_t)(x0 + ty + i) * DIM + (y0 + tx)] = (__bf16)t[tx][ty + i];
}

// ---------------------------------------------------------------------------
// KNN: per query, top-8 nearest anchors (anchors cached in LDS)
// ---------------------------------------------------------------------------
__global__ __launch_bounds__(256) void knn_kernel(
    const float* __restrict__ xyz_q, const float* __restrict__ anchors_xyz,
    int* __restrict__ knn_idx)
{
  __shared__ float ax[NA], ay[NA], az[NA];
  const int b = blockIdx.x / (NQ / 256);
  const int q = (blockIdx.x % (NQ / 256)) * 256 + threadIdx.x;
  const float* A = anchors_xyz + (size_t)b * NA * 3;
  for (int i = threadIdx.x; i < NA; i += 256) {
    ax[i] = A[i * 3 + 0];
    ay[i] = A[i * 3 + 1];
    az[i] = A[i * 3 + 2];
  }
  __syncthreads();

  const float qx = xyz_q[((size_t)b * NQ + q) * 3 + 0];
  const float qy = xyz_q[((size_t)b * NQ + q) * 3 + 1];
  const float qz = xyz_q[((size_t)b * NQ + q) * 3 + 2];

  float bd[KNN];
  int   bi[KNN];
  #pragma unroll
  for (int j = 0; j < KNN; ++j) { bd[j] = 3.4e38f; bi[j] = 0; }

  for (int n = 0; n < NA; ++n) {
    const float dx = qx - ax[n], dy = qy - ay[n], dz = qz - az[n];
    const float d  = dx * dx + dy * dy + dz * dz;
    if (d < bd[KNN - 1]) {
      bd[KNN - 1] = d; bi[KNN - 1] = n;
      #pragma unroll
      for (int j = KNN - 1; j > 0; --j) {
        if (bd[j] < bd[j - 1]) {
          float td = bd[j]; bd[j] = bd[j - 1]; bd[j - 1] = td;
          int   ti = bi[j]; bi[j] = bi[j - 1]; bi[j - 1] = ti;
        }
      }
    }
  }
  int* out = knn_idx + ((size_t)b * NQ + q) * KNN;
  #pragma unroll
  for (int j = 0; j < KNN; ++j) out[j] = bi[j];
}

// ---------------------------------------------------------------------------
// q_attn / kg / vg from global_feats (tiny matvecs, B=2)
// qg layout: [q_attn(B,DIM) | kg(B,DIM) | vg(B,DIM)]
// ---------------------------------------------------------------------------
__global__ __launch_bounds__(256) void global_proj_kernel(
    const float* __restrict__ gf, const float* __restrict__ wqs,
    const float* __restrict__ wkg, const float* __restrict__ wvg,
    float* __restrict__ qg)
{
  const int t = blockIdx.x * 256 + threadIdx.x;
  if (t >= B_ * DIM) return;
  const int b = t / DIM, c = t % DIM;
  const float* g = gf + (size_t)b * DIM;
  float sq = 0.f, sk = 0.f, sv = 0.f;
  for (int k = 0; k < DIM; ++k) {
    const float gv = g[k];
    sq += gv * wqs[(size_t)k * DIM + c];
    sk += gv * wkg[(size_t)k * DIM + c];
    sv += gv * wvg[(size_t)k * DIM + c];
  }
  qg[(size_t)b * DIM + c]            = sq;
  qg[(size_t)(B_ + b) * DIM + c]     = sk;
  qg[(size_t)(2 * B_ + b) * DIM + c] = sv;
}

// ---------------------------------------------------------------------------
// h1_pe = relu(d @ d1_w + d1_b), d = xyz_q - anchors_xyz[knn]   (K-dim = 3)
// ---------------------------------------------------------------------------
__global__ __launch_bounds__(256) void pe_h1_kernel(
    const float* __restrict__ xyz_q, const float* __restrict__ anchors_xyz,
    const int* __restrict__ knn_idx, const float* __restrict__ d1w,
    const float* __restrict__ d1b, __bf16* __restrict__ h1)
{
  const int row = blockIdx.x;          // (b*NQ+q)*KNN + j
  const int bq  = row / KNN;
  const int j   = row % KNN;
  const int b   = bq / NQ;
  const int a   = knn_idx[(size_t)bq * KNN + j];
  const float* qp = xyz_q + (size_t)bq * 3;
  const float* ap = anchors_xyz + ((size_t)b * NA + a) * 3;
  const float dx = qp[0] - ap[0], dy = qp[1] - ap[1], dz = qp[2] - ap[2];
  for (int c = threadIdx.x; c < DIM; c += 256) {
    const float v = dx * d1w[c] + dy * d1w[DIM + c] + dz * d1w[2 * DIM + c] + d1b[c];
    h1[(size_t)row * DIM + c] = (__bf16)fmaxf(v, 0.0f);
  }
}

// ---------------------------------------------------------------------------
// attn_in[b,q,j,:] = q_attn - k_attn(j) + pos(j)   (bf16 GEMM operand)
// ---------------------------------------------------------------------------
__global__ __launch_bounds__(256) void attn_in_kernel(
    const float* __restrict__ qg, const float* __restrict__ k_anchor,
    const __bf16* __restrict__ pe, const int* __restrict__ knn_idx,
    __bf16* __restrict__ attn_in)
{
  const int row = blockIdx.x;          // (b*NQ+q)*9 + j
  const int bq  = row / 9;
  const int j   = row % 9;
  const int b   = bq / NQ;
  const float* qv  = qg + (size_t)b * DIM;          // q_attn
  const float* kgv = qg + (size_t)(B_ + b) * DIM;   // kg
  if (j < KNN) {
    const int a = knn_idx[(size_t)bq * KNN + j];
    const float*  kr = k_anchor + ((size_t)b * NA + a) * DIM;
    const __bf16* pr = pe + ((size_t)bq * KNN + j) * DIM;
    for (int c = threadIdx.x; c < DIM; c += 256)
      attn_in[(size_t)row * DIM + c] = (__bf16)(qv[c] - kr[c] + (float)pr[c]);
  } else {
    for (int c = threadIdx.x; c < DIM; c += 256)
      attn_in[(size_t)row * DIM + c] = (__bf16)(qv[c] - kgv[c]);
  }
}

// ---------------------------------------------------------------------------
// softmax over the 9-token axis per (query, channel); lat = sum attn*(v+pos)
// ---------------------------------------------------------------------------
__global__ __launch_bounds__(256) void softmax_lat_kernel(
    const float* __restrict__ logits, const float* __restrict__ v_anchor,
    const float* __restrict__ qg, const __bf16* __restrict__ pe,
    const int* __restrict__ knn_idx, __bf16* __restrict__ lat)
{
  const int bq = blockIdx.x;           // b*NQ + q
  const int b  = bq / NQ;
  __shared__ int sidx[KNN];
  if (threadIdx.x < KNN) sidx[threadIdx.x] = knn_idx[(size_t)bq * KNN + threadIdx.x];
  __syncthreads();
  const float* L  = logits + (size_t)bq * 9 * DIM;
  const float* vg = qg + (size_t)(2 * B_ + b) * DIM;
  for (int c = threadIdx.x; c < DIM; c += 256) {
    float l[9];
    float mx = -3.4e38f;
    #pragma unroll
    for (int j = 0; j < 9; ++j) { l[j] = L[(size_t)j * DIM + c]; mx = fmaxf(mx, l[j]); }
    float s = 0.f;
    #pragma unroll
    for (int j = 0; j < 9; ++j) { l[j] = __expf(l[j] - mx); s += l[j]; }
    const float inv = 1.0f / s;
    float acc = 0.f;
    #pragma unroll
    for (int j = 0; j < KNN; ++j) {
      const float vv = v_anchor[((size_t)b * NA + sidx[j]) * DIM + c]
                     + (float)pe[((size_t)bq * KNN + j) * DIM + c];
      acc += l[j] * inv * vv;
    }
    acc += l[8] * inv * vg[c];
    lat[(size_t)bq * DIM + c] = (__bf16)acc;
  }
}

// ---------------------------------------------------------------------------
// net0 = PE(xyz_q) @ fc_p_w + fc_p_b  (PE staged in LDS, 60-dim contraction)
// ---------------------------------------------------------------------------
__global__ __launch_bounds__(256) void decoder_input_kernel(
    const float* __restrict__ xyz_q, const float* __restrict__ fcpw,
    const float* __restrict__ fcpb, float* __restrict__ net)
{
  const int row = blockIdx.x;          // b*NQ + q
  __shared__ float pe[60];
  if (threadIdx.x < 60) {
    const int t = threadIdx.x;
    const int f = t / 6, rem = t % 6, s = rem / 3, d = rem % 3;
    const float p = xyz_q[(size_t)row * 3 + d];
    const float a = p * exp2f((float)f) * 3.14159265358979323846f;
    pe[t] = s ? cosf(a) : sinf(a);
  }
  __syncthreads();
  for (int c = threadIdx.x; c < HID; c += 256) {
    float acc = fcpb[c];
    #pragma unroll
    for (int k = 0; k < 60; ++k) acc += pe[k] * fcpw[(size_t)k * HID + c];
    net[(size_t)row * HID + c] = acc;
  }
}

// ---------------------------------------------------------------------------
// Launch pipeline
// ---------------------------------------------------------------------------
extern "C" void kernel_launch(void* const* d_in, const int* in_sizes, int n_in,
                              void* d_out, int out_size, void* d_ws, size_t ws_size,
                              hipStream_t stream)
{
  (void)in_sizes; (void)n_in; (void)out_size; (void)ws_size;
  const float* xyz_q         = (const float*)d_in[0];
  const float* global_feats  = (const float*)d_in[1];
  const float* anchors_xyz   = (const float*)d_in[2];
  const float* anchors_feats = (const float*)d_in[3];
  const float* w_qs = (const float*)d_in[4];
  const float* w_ks = (const float*)d_in[5];
  const float* w_vs = (const float*)d_in[6];
  const float* w_kg = (const float*)d_in[7];
  const float* w_vg = (const float*)d_in[8];
  const float* d1_w = (const float*)d_in[9];
  const float* d1_b = (const float*)d_in[10];
  const float* d2_w = (const float*)d_in[11];
  const float* d2_b = (const float*)d_in[12];
  const float* g1_w = (const float*)d_in[13];
  const float* g1_b = (const float*)d_in[14];
  const float* g2_w = (const float*)d_in[15];
  const float* g2_b = (const float*)d_in[16];
  const float* fc_p_w = (const float*)d_in[17];
  const float* fc_p_b = (const float*)d_in[18];
  const float* fc_c_w = (const float*)d_in[19];
  const float* fc_c_b = (const float*)d_in[20];
  const float* blk0_w = (const float*)d_in[21];
  const float* blk0_b = (const float*)d_in[22];
  const float* blk1_w = (const float*)d_in[23];
  const float* blk1_b = (const float*)d_in[24];

  char* ws = (char*)d_ws;
  size_t off = 0;
  auto alloc = [&](size_t bytes) -> void* {
    off = (off + 255) & ~(size_t)255;
    void* p = ws + off;
    off += bytes;
    return p;
  };
  constexpr size_t WSZ = (size_t)DIM * DIM;   // elements per weight matrix
  int*    knn_idx  = (int*)   alloc((size_t)B_ * NQ * KNN * 4);
  float*  qg       = (float*) alloc((size_t)3 * B_ * DIM * 4);
  __bf16* bt       = (__bf16*)alloc(20 * WSZ * 2);           // bf16 [N,K] weights
  float*  k_anchor = (float*) alloc((size_t)B_ * NA * DIM * 4);
  float*  v_anchor = (float*) alloc((size_t)B_ * NA * DIM * 4);
  __bf16* h1_pe    = (__bf16*)alloc((size_t)B_ * NQ * KNN * DIM * 2);
  __bf16* pe_loc   = (__bf16*)alloc((size_t)B_ * NQ * KNN * DIM * 2);
  __bf16* attn_in  = (__bf16*)alloc((size_t)B_ * NQ * 9 * DIM * 2);
  __bf16* h2       = (__bf16*)alloc((size_t)B_ * NQ * 9 * DIM * 2);
  float*  logits   = (float*) alloc((size_t)B_ * NQ * 9 * DIM * 4);
  __bf16* lat      = (__bf16*)alloc((size_t)B_ * NQ * DIM * 2);
  float*  net      = (float*) alloc((size_t)B_ * NQ * HID * 4);
  __bf16* hdec     = (__bf16*)alloc((size_t)B_ * NQ * HID * 2);

  const dim3 blk(256);

  // 0) one-time weight convert+transpose -> bf16 [N,K]
  //    slots: 0 w_ks | 1 w_vs | 2 d2_w | 3 g1_w | 4 g2_w | 5..9 fc_c | 10..14 blk0 | 15..19 blk1
  wtrans_kernel<<<dim3(16, 16, 1), blk, 0, stream>>>(w_ks, bt + 0 * WSZ);
  wtrans_kernel<<<dim3(16, 16, 1), blk, 0, stream>>>(w_vs, bt + 1 * WSZ);
  wtrans_kernel<<<dim3(16, 16, 1), blk, 0, stream>>>(d2_w, bt + 2 * WSZ);
  wtrans_kernel<<<dim3(16, 16, 1), blk, 0, stream>>>(g1_w, bt + 3 * WSZ);
  wtrans_kernel<<<dim3(16, 16, 1), blk, 0, stream>>>(g2_w, bt + 4 * WSZ);
  wtrans_kernel<<<dim3(16, 16, 5), blk, 0, stream>>>(fc_c_w, bt + 5 * WSZ);
  wtrans_kernel<<<dim3(16, 16, 5), blk, 0, stream>>>(blk0_w, bt + 10 * WSZ);
  wtrans_kernel<<<dim3(16, 16, 5), blk, 0, stream>>>(blk1_w, bt + 15 * WSZ);

  // 1) KNN
  knn_kernel<<<B_ * (NQ / 256), blk, 0, stream>>>(xyz_q, anchors_xyz, knn_idx);
  // 2) global-token projections
  global_proj_kernel<<<(B_ * DIM + 255) / 256, blk, 0, stream>>>(global_feats, w_qs, w_kg, w_vg, qg);
  // 3) anchor K/V projections (WMMA + TDM B-panel)
  {
    dim3 g(DIM / BN, (B_ * NA) / BM);
    gemm_bf16_wmma<0, false, false, false><<<g, blk, 0, stream>>>(
        anchors_feats, bt + 0 * WSZ, nullptr, nullptr, k_anchor, nullptr, B_ * NA, DIM, DIM);
    gemm_bf16_wmma<0, false, false, false><<<g, blk, 0, stream>>>(
        anchors_feats, bt + 1 * WSZ, nullptr, nullptr, v_anchor, nullptr, B_ * NA, DIM, DIM);
  }
  // 4) fc_delta layer 1 (K=3, VALU) then layer 2 (WMMA) -> pos (bf16)
  pe_h1_kernel<<<B_ * NQ * KNN, blk, 0, stream>>>(xyz_q, anchors_xyz, knn_idx, d1_w, d1_b, h1_pe);
  {
    dim3 g(DIM / BN, (B_ * NQ * KNN) / BM);
    gemm_bf16_wmma<2, false, false, true><<<g, blk, 0, stream>>>(
        h1_pe, bt + 2 * WSZ, d2_b, nullptr, nullptr, pe_loc, B_ * NQ * KNN, DIM, DIM);
  }
  // 5) attn_in = q - k + pos (gather + elementwise -> bf16)
  attn_in_kernel<<<B_ * NQ * 9, blk, 0, stream>>>(qg, k_anchor, pe_loc, knn_idx, attn_in);
  // 6) fc_gamma: two WMMA GEMMs (fused ReLU on first)
  {
    dim3 g(DIM / BN, (B_ * NQ * 9) / BM);
    gemm_bf16_wmma<2, false, true, true><<<g, blk, 0, stream>>>(
        attn_in, bt + 3 * WSZ, g1_b, nullptr, nullptr, h2, B_ * NQ * 9, DIM, DIM);
    gemm_bf16_wmma<2, false, false, false><<<g, blk, 0, stream>>>(
        h2, bt + 4 * WSZ, g2_b, nullptr, logits, nullptr, B_ * NQ * 9, DIM, DIM);
  }
  // 7) softmax over neighbor axis + weighted sum -> lat (bf16)
  softmax_lat_kernel<<<B_ * NQ, blk, 0, stream>>>(logits, v_anchor, qg, pe_loc, knn_idx, lat);
  // 8) decoder input: net0 = PE(xyz) @ fc_p_w + b
  decoder_input_kernel<<<B_ * NQ, blk, 0, stream>>>(xyz_q, fc_p_w, fc_p_b, net);
  // 9) 5 ResNet blocks: three WMMA GEMMs each, residuals fused in epilogue
  {
    dim3 g(HID / BN, (B_ * NQ) / BM);
    for (int i = 0; i < NBLK; ++i) {
      // net += lat @ fc_c_w[i] + fc_c_b[i]
      gemm_bf16_wmma<2, true, false, false><<<g, blk, 0, stream>>>(
          lat, bt + (5 + i) * WSZ, fc_c_b + (size_t)i * HID,
          net, net, nullptr, B_ * NQ, HID, DIM);
      // hdec = relu( relu(net) @ blk0_w[i] + blk0_b[i] )   (ReLU-on-A fused)
      gemm_bf16_wmma<1, false, true, true><<<g, blk, 0, stream>>>(
          net, bt + (10 + i) * WSZ, blk0_b + (size_t)i * HID,
          nullptr, nullptr, hdec, B_ * NQ, HID, HID);
      // net += hdec @ blk1_w[i] + blk1_b[i]; final block writes d_out
      float* outp = (i == NBLK - 1) ? (float*)d_out : net;
      gemm_bf16_wmma<2, true, false, false><<<g, blk, 0, stream>>>(
          hdec, bt + (15 + i) * WSZ, blk1_b + (size_t)i * HID,
          net, outp, nullptr, B_ * NQ, HID, HID);
    }
  }
}
